// graph_sep_init_63651415327271
// MI455X (gfx1250) — matrix operand order repaired and checked by reading, not compiled, not run on hardware
//
#include <hip/hip_runtime.h>

// ---------------- problem constants ----------------
#define N_MOD   2
#define N_SUB   4096
#define DIM     64
#define TOPK    20
#define NTOT    (N_MOD * N_SUB)      // 8192
#define NMAT    (N_MOD * N_MOD)      // 4
#define ALPHA   3.0f
#define NEG_INF (-3.402823466e+38f)

#define CSPLIT       4               // column quarters for partial top-k
#define COLS_PER_WG  (NTOT / CSPLIT) // 2048
#define ROWS_PER_WG  128
#define NTILES       (COLS_PER_WG / 16)   // 128 column tiles per WG

// ---------------- CDNA5 WMMA vector types ----------------
typedef __attribute__((ext_vector_type(16))) _Float16 v16h;
typedef __attribute__((ext_vector_type(8)))  _Float16 v8h;
typedef __attribute__((ext_vector_type(8)))  float    v8f;

#define AS3 __attribute__((address_space(3)))

// raw LDS byte offset of a __shared__ object (generic -> AS3 addrspacecast)
static __device__ __forceinline__ unsigned lds_offset(const void* p) {
  return (unsigned)(unsigned long long)(AS3 const char*)p;
}

// CDNA5 async copy: LDS[dst] = MEM[src], 16 bytes per lane, tracked by ASYNCcnt
static __device__ __forceinline__ void async_copy_b128(unsigned lds_byte_off,
                                                       const void* gptr) {
  asm volatile("global_load_async_to_lds_b128 %0, %1, off"
               :: "v"(lds_byte_off), "v"(gptr) : "memory");
}

static __device__ __forceinline__ void wait_asynccnt0() {
#if __has_builtin(__builtin_amdgcn_s_wait_asynccnt)
  __builtin_amdgcn_s_wait_asynccnt(0);
#else
  asm volatile("s_wait_asynccnt 0" ::: "memory");
#endif
}

// gfx1250 has V_TANH_F32; use the hardware transcendental when clang exposes it.
static __device__ __forceinline__ float hw_tanhf(float x) {
#if __has_builtin(__builtin_amdgcn_tanhf)
  return __builtin_amdgcn_tanhf(x);
#else
  return tanhf(x);
#endif
}

// A fragment (16x32 f16, ISA 7.12.2): lane&15 = row M, lane>>4 selects K-half.
// per-lane halves: [0..7] = K base+0..7, [8..15] = K base+16..23 (base = (lane>>4)*8)
static __device__ __forceinline__ v16h ldA(const _Float16* p) {
  v8h lo = *(const v8h*)(p);        // 16B
  v8h hi = *(const v8h*)(p + 16);   // 16B
  v16h r;
#pragma unroll
  for (int e = 0; e < 8; ++e) { r[e] = lo[e]; r[e + 8] = hi[e]; }
  return r;
}

// ---------------- kernel 0: zero the 256MB output ----------------
__global__ void __launch_bounds__(256) zero_kernel(float4* __restrict__ out) {
  size_t i = (size_t)blockIdx.x * 256 + threadIdx.x;
  out[i] = make_float4(0.f, 0.f, 0.f, 0.f);
}

// ---------------- kernel 1: v = emb @ w + b, store f16 ----------------
// transposeOut=0: v[k][n][e] row-major (WMMA A operand, v1)
// transposeOut=1: v[k][e][n] (K-major, WMMA B operand, v2^T)
__global__ void __launch_bounds__(256)
linear_kernel(const float* __restrict__ emb, const float* __restrict__ w,
              const float* __restrict__ bias, _Float16* __restrict__ outp,
              int transposeOut) {
  __shared__ float wl[DIM * DIM];
  __shared__ float el[64 * DIM];
  __shared__ float bl[DIM];
  const int k  = blockIdx.y;
  const int n0 = blockIdx.x * 64;
  const int t  = threadIdx.x;
#pragma unroll
  for (int i = 0; i < 16; ++i) {
    int idx = t + 256 * i;
    wl[idx] = w[(size_t)k * DIM * DIM + idx];
    el[idx] = emb[((size_t)k * N_SUB + n0) * DIM + idx];
  }
  if (t < DIM) bl[t] = bias[k * DIM + t];
  __syncthreads();
#pragma unroll 4
  for (int o = 0; o < 16; ++o) {
    int idx = t + 256 * o;
    int row = idx >> 6, col = idx & 63;
    float acc = bl[col];
#pragma unroll
    for (int d = 0; d < DIM; ++d)
      acc = fmaf(el[row * DIM + d], wl[d * DIM + col], acc);
    size_t off = transposeOut ? ((size_t)col * N_SUB + (n0 + row))
                              : ((size_t)(n0 + row) * DIM + col);
    outp[(size_t)k * N_SUB * DIM + off] = (_Float16)acc;
  }
}

// ---------------- kernel 2: fused WMMA GEMM + partial top-k on RAW scores --
// relu(tanh(alpha*x)) is monotone non-decreasing => top-k on raw scores is
// equivalent (ties at <=0 all produce output value 0); activation deferred to
// the merge kernel. B tiles are staged into LDS with the CDNA5 async copy
// engine (double-buffered, one workgroup barrier per tile covers both the
// staged B tile for ct+1 and the score tile for ct).
// grid (64 row-stripes, CSPLIT col-quarters), 128 threads = 4 waves.
__global__ void __launch_bounds__(128)
score_topk_partial(const _Float16* __restrict__ v1h,
                   const _Float16* __restrict__ v2t,
                   float* __restrict__ partv, int* __restrict__ partc) {
  __shared__ _Float16 bstage[2][64 * 16];     // 2 x 2KB B tiles (K=64 x N=16)
  __shared__ float sc[2][ROWS_PER_WG * 16];   // ping-pong raw-score tiles
  __shared__ float topv[ROWS_PER_WG * TOPK];
  __shared__ int   topc[ROWS_PER_WG * TOPK];

  const int tid  = threadIdx.x;
  const int lane = tid & 31;
  const int wave = tid >> 5;
  const int r0   = blockIdx.x * ROWS_PER_WG;
  const int cs   = blockIdx.y;
  const int i    = r0 / N_SUB;
  const int j    = cs >> 1;
  const int k    = i * N_MOD + j;                 // block index
  const int cbase = cs * COLS_PER_WG;             // absolute column base
  const int cloc0 = (cs & 1) * COLS_PER_WG;       // column base inside block j

#pragma unroll
  for (int e = 0; e < TOPK; ++e) {
    topv[tid * TOPK + e] = NEG_INF;
    topc[tid * TOPK + e] = 0;
  }
  float minv = NEG_INF;
  int   minIdx = 0;

  // A fragments: this wave owns rows r0 + wave*32 .. +31 (two 16-row tiles),
  // K = 64 split into two 16x32 fragments each. Loaded once, reused 128x.
  const int rowA = r0 + wave * 32 + (lane & 15);
  const _Float16* ap0 = v1h + ((size_t)k * N_SUB + rowA) * DIM + ((lane >> 4) << 3);
  const _Float16* ap1 = ap0 + (size_t)16 * DIM;
  v16h a[2][2];
#pragma unroll
  for (int kc = 0; kc < 2; ++kc) {
    a[0][kc] = ldA(ap0 + kc * 32);
    a[1][kc] = ldA(ap1 + kc * 32);
  }

  // async B staging: thread t copies 16B of row (t>>1): tile is 64 rows x 32B
  const int      trow  = tid >> 1;                 // 0..63 = K row
  const int      tpart = (tid & 1) * 8;            // half-row in f16 units
  const _Float16* gB   = v2t + ((size_t)k * DIM + trow) * N_SUB + cloc0 + tpart;
  const unsigned ldsS0 = lds_offset(&bstage[0][trow * 16 + tpart]);
  const unsigned ldsS1 = lds_offset(&bstage[1][trow * 16 + tpart]);

  // prologue: stage tile 0 into buffer 0
  async_copy_b128(ldsS0, gB);
  wait_asynccnt0();
  __syncthreads();

  for (int ct = 0; ct < NTILES; ++ct) {
    // stage tile ct+1 into the other buffer (overlaps with WMMA below)
    if (ct + 1 < NTILES)
      async_copy_b128(((ct + 1) & 1) ? ldsS1 : ldsS0, gB + (ct + 1) * 16);
    __builtin_prefetch(gB + (ct + 2) * 16, 0, 2);  // global_prefetch_b8, 2 ahead

    // B fragments from LDS: lane&15 = N column, elements over K in A's order
    const _Float16* bs = &bstage[ct & 1][0];
    v16h b0, b1;
#pragma unroll
    for (int s = 0; s < 16; ++s) {
      int kk = ((s & 8) << 1) + ((lane >> 4) << 3) + (s & 7);
      b0[s] = bs[kk * 16 + (lane & 15)];
      b1[s] = bs[(32 + kk) * 16 + (lane & 15)];
    }

#pragma unroll
    for (int rt = 0; rt < 2; ++rt) {
      v8f c = {0.f, 0.f, 0.f, 0.f, 0.f, 0.f, 0.f, 0.f};
      c = __builtin_amdgcn_wmma_f32_16x16x32_f16(false, a[rt][0], false, b0,
                                                 (short)0, c, false, false);
      c = __builtin_amdgcn_wmma_f32_16x16x32_f16(false, a[rt][1], false, b1,
                                                 (short)0, c, false, false);
      // raw scores to LDS (C layout: M = e + 8*(lane>=16), N = lane&15)
#pragma unroll
      for (int e = 0; e < 8; ++e) {
        int rowl = wave * 32 + rt * 16 + e + ((lane >> 4) << 3);
        int coll = lane & 15;
        sc[ct & 1][rowl * 16 + coll] = c[e];
      }
    }
    wait_asynccnt0();   // tile ct+1 resident before the barrier releases
    __syncthreads();    // publishes sc[ct&1] and bstage[(ct+1)&1]

    // each thread owns one row; insert 16 raw candidates into its top-20 list
    const int colg0 = cbase + ct * 16;
#pragma unroll
    for (int cc = 0; cc < 16; ++cc) {
      float v = sc[ct & 1][tid * 16 + cc];
      if (v > minv) {
        topv[tid * TOPK + minIdx] = v;
        topc[tid * TOPK + minIdx] = colg0 + cc;
        minv = topv[tid * TOPK]; minIdx = 0;
#pragma unroll
        for (int e = 1; e < TOPK; ++e) {
          float tv = topv[tid * TOPK + e];
          if (tv < minv) { minv = tv; minIdx = e; }
        }
      }
    }
  }

  // emit partial lists (raw scores)
  const int r = r0 + tid;
#pragma unroll
  for (int e = 0; e < TOPK; ++e) {
    partv[((size_t)r * CSPLIT + cs) * TOPK + e] = topv[tid * TOPK + e];
    partc[((size_t)r * CSPLIT + cs) * TOPK + e] = topc[tid * TOPK + e];
  }
}

// ---------------- kernel 3: merge partials + deferred activation ----------
__global__ void __launch_bounds__(256)
topk_merge(const float* __restrict__ partv, const int* __restrict__ partc,
           float* __restrict__ out) {
  __shared__ float tv[256 * TOPK];
  __shared__ int   tc[256 * TOPK];
  const int t = threadIdx.x;
  const int r = blockIdx.x * 256 + t;
#pragma unroll
  for (int e = 0; e < TOPK; ++e) { tv[t * TOPK + e] = NEG_INF; tc[t * TOPK + e] = 0; }
  float minv = NEG_INF;
  int   minIdx = 0;
  for (int cs = 0; cs < CSPLIT; ++cs) {
#pragma unroll
    for (int e = 0; e < TOPK; ++e) {
      float v = partv[((size_t)r * CSPLIT + cs) * TOPK + e];
      int   c = partc[((size_t)r * CSPLIT + cs) * TOPK + e];
      if (v > minv) {
        tv[t * TOPK + minIdx] = v;
        tc[t * TOPK + minIdx] = c;
        minv = tv[t * TOPK]; minIdx = 0;
#pragma unroll
        for (int e2 = 1; e2 < TOPK; ++e2) {
          float x = tv[t * TOPK + e2];
          if (x < minv) { minv = x; minIdx = e2; }
        }
      }
    }
  }
  // deferred activation: only positive raw scores produce nonzero output;
  // zeros are already present from the zero-fill pass.
#pragma unroll
  for (int e = 0; e < TOPK; ++e) {
    float raw = tv[t * TOPK + e];
    if (raw > 0.f) {
      float s = hw_tanhf(ALPHA * raw);          // v_tanh_f32 on gfx1250
      out[(size_t)r * NTOT + tc[t * TOPK + e]] = s;
    }
  }
}

// ---------------- host launcher ----------------
extern "C" void kernel_launch(void* const* d_in, const int* in_sizes, int n_in,
                              void* d_out, int out_size, void* d_ws, size_t ws_size,
                              hipStream_t stream) {
  (void)in_sizes; (void)n_in; (void)out_size; (void)ws_size;
  // inputs: 0=idx (unused), 1=emb1, 2=emb2, 3=w1, 4=w2, 5=b1, 6=b2
  const float* emb1 = (const float*)d_in[1];
  const float* emb2 = (const float*)d_in[2];
  const float* w1   = (const float*)d_in[3];
  const float* w2   = (const float*)d_in[4];
  const float* b1   = (const float*)d_in[5];
  const float* b2   = (const float*)d_in[6];
  float* out = (float*)d_out;

  // workspace layout
  char* ws = (char*)d_ws;
  const size_t vbytes = (size_t)NMAT * N_SUB * DIM * sizeof(_Float16); // 2 MB each
  _Float16* v1h = (_Float16*)ws;                    // row-major f16
  _Float16* v2t = (_Float16*)(ws + vbytes);         // transposed f16 [k][d][m]
  float* partv  = (float*)(ws + 2 * vbytes);
  int*   partc  = (int*)(ws + 2 * vbytes + (size_t)NTOT * CSPLIT * TOPK * sizeof(float));

  // 1) zero 256MB output (scatter targets only top-k positions)
  zero_kernel<<<(size_t)NTOT * NTOT / 4 / 256, 256, 0, stream>>>((float4*)out);

  // 2) per-block linears -> f16 operands
  dim3 gl(N_SUB / 64, NMAT);
  linear_kernel<<<gl, 256, 0, stream>>>(emb1, w1, b1, v1h, 0);
  linear_kernel<<<gl, 256, 0, stream>>>(emb2, w2, b2, v2t, 1);

  // 3) fused WMMA GEMM + partial top-k on raw scores (async B staging)
  dim3 gs(NTOT / ROWS_PER_WG, CSPLIT);
  score_topk_partial<<<gs, 128, 0, stream>>>(v1h, v2t, partv, partc);

  // 4) merge partials, apply activation, scatter into output
  topk_merge<<<NTOT / 256, 256, 0, stream>>>(partv, partc, out);
}
